// ClassificationHead_53206054863029
// MI455X (gfx1250) — matrix operand order repaired and checked by reading, not compiled, not run on hardware
//
#include <hip/hip_runtime.h>
#include <hip/hip_bf16.h>
#include <math.h>

// ---------------- problem constants ----------------
#define BB 8
#define QQ 64
#define BQ 512          // B*Q
#define CD 512          // C
#define HWN 256         // H*W
#define EE 512          // E
#define HATT 128
#define HMLP 512
#define NCLS 14

typedef __attribute__((ext_vector_type(16))) _Float16 v16h;
typedef __attribute__((ext_vector_type(8)))  _Float16 v8h;
typedef __attribute__((ext_vector_type(8)))  float    v8f;
typedef __attribute__((ext_vector_type(4)))  unsigned int u32x4;
typedef __attribute__((ext_vector_type(8)))  int i32x8;
typedef __attribute__((ext_vector_type(4)))  int i32x4;

#if defined(__has_builtin)
#if __has_builtin(__builtin_amdgcn_tensor_load_to_lds)
#define HAVE_TDM 1
#endif
#endif

__device__ __forceinline__ float gelu_exact(float x) {
    return 0.5f * x * (1.f + erff(x * 0.70710678118654752f));
}

// ---------------- kernel 0: transpose + f32->f16 convert ----------------
// src: (batch, R, Cc) row-major f32  ->  dst: (batch, Cc, R) row-major f16
__global__ void tcvt_kernel(const float* __restrict__ src, _Float16* __restrict__ dst,
                            int batch, int R, int Cc) {
    int total = batch * R * Cc;
    for (int i = blockIdx.x * blockDim.x + threadIdx.x; i < total;
         i += gridDim.x * blockDim.x) {
        int b   = i / (R * Cc);
        int rem = i - b * R * Cc;
        int r   = rem / Cc;
        int c   = rem - r * Cc;
        dst[b * R * Cc + c * R + r] = (_Float16)src[i];
    }
}

// ---------------- generic WMMA GEMM: C = op(A_f32 (opt. gathered) x BT_f16) + bias ---
// A: M x K f32 (row stride lda). If gather != nullptr, row i of A is A + gather[i]*lda.
// BT: N x K f16 (row-major, stride K)  == B transposed, so B-fragments are contiguous.
// C: M x N f32 (row stride ldc). One wave per 16x16 tile; 8 waves / 256-thr block.
__global__ __launch_bounds__(256) void gemm_f32a_f16b(
    const float* __restrict__ A, int lda,
    const int* __restrict__ gather,
    const _Float16* __restrict__ BT, int K,
    const float* __restrict__ bias,
    float* __restrict__ C, int ldc,
    int ntN, int gelu_flag)
{
    const int lane = threadIdx.x & 31;
    const int l16  = lane & 15;
    const int hi   = lane >> 4;
    const int wg   = blockIdx.x * 8 + (threadIdx.x >> 5);
    const int mt   = wg / ntN;
    const int nt   = wg - mt * ntN;
    const int m0   = mt << 4;
    const int n0   = nt << 4;

    const int rowid = m0 + l16;
    const int arow  = gather ? gather[rowid] : rowid;
    const float* ap = A + (long)arow * lda;

    v8f acc = {};
    for (int k0 = 0; k0 < K; k0 += 32) {
        // A fragment: 16x32 f16, quarter-interleaved layout
        const float* pa = ap + k0 + hi * 8;
        v16h af;
#pragma unroll
        for (int j = 0; j < 8; ++j) {
            af[j]     = (_Float16)pa[j];
            af[8 + j] = (_Float16)pa[16 + j];
        }
        // B fragment: quarter-interleaved layout (same k order as A)
        const _Float16* pb = BT + (long)(n0 + l16) * K + k0 + hi * 8;
        v8h blo = *(const v8h*)pb;
        v8h bhi = *(const v8h*)(pb + 16);
        v16h bf;
#pragma unroll
        for (int j = 0; j < 8; ++j) { bf[j] = blo[j]; bf[8 + j] = bhi[j]; }

        acc = __builtin_amdgcn_wmma_f32_16x16x32_f16(false, af, false, bf,
                                                     (short)0, acc, false, false);
    }
    const int col = n0 + l16;
    const float bv = bias[col];
#pragma unroll
    for (int r = 0; r < 8; ++r) {
        float v = acc[r] + bv;
        if (gelu_flag) v = gelu_exact(v);
        C[(long)(m0 + r + hi * 8) * ldc + col] = v;
    }
}

// ---------------- big kernel: per (b,q) FiLM-attention + pooling ----------------
// hidden = (fT(b) .* gamma) @ W_att_h + cst ; attn = sigmoid(gelu(hidden) @ Wf + b)
// pooled = gamma .* (attn^T f)/S' + beta * S/S'
// W_att_h^T chunks are staged into LDS by the Tensor Data Mover (padded tile).
__global__ __launch_bounds__(512) void attn_pool_kernel(
    const _Float16* __restrict__ fT,      // (B, HW, C) f16 row-major
    const _Float16* __restrict__ WattT,   // (HATT, C) f16
    const float* __restrict__ feature,    // (B, C, HW) f32 (original layout)
    const float* __restrict__ gb,         // (BQ, 2C): gamma | beta
    const float* __restrict__ cst,        // (BQ, HATT)
    const float* __restrict__ W_att_f,    // (HATT)
    const float* __restrict__ b_att_f,    // (1)
    float* __restrict__ pooled)           // (BQ, C)
{
    __shared__ __align__(16) _Float16 WqT[HATT * 136];   // 136 = 128 + 8 pad (272B rows)
    __shared__ __align__(16) _Float16 gammaH[CD];
    __shared__ float gammaLds[CD];
    __shared__ float betaLds[CD];
    __shared__ float attnLds[HWN];
    __shared__ float wfLds[HATT];
    __shared__ float Ssum;

    const int bq   = blockIdx.x;
    const int b    = bq >> 6;
    const int tid  = threadIdx.x;
    const int lane = tid & 31;
    const int l16  = lane & 15;
    const int hi   = lane >> 4;
    const int wid  = tid >> 5;      // 0..15 : M tile (16 rows of HW)
    const int m0   = wid << 4;

    if (tid < CD) {
        float g = gb[bq * 1024 + tid];
        gammaLds[tid] = g;
        gammaH[tid]   = (_Float16)g;
        betaLds[tid]  = gb[bq * 1024 + 512 + tid];
    }
    if (tid < HATT) wfLds[tid] = W_att_f[tid];
    if (tid == 0)   Ssum = 0.f;

    v8f acc[8] = {};
    const _Float16* fTb = fT + b * (HWN * CD);

    for (int kc = 0; kc < 4; ++kc) {
        __syncthreads();
#if defined(HAVE_TDM)
        // Tensor Data Mover: copy 128(h) x 128(c) f16 tile of WattT into LDS with
        // 16B row padding (pad_interval=5 -> every 64 DWORDs, pad_amount=3 -> 4 DWORDs).
        if (wid == 0) {
            unsigned lds_base = (unsigned)(unsigned long long)(void*)WqT;
            unsigned long long ga =
                (unsigned long long)(const void*)(WattT + (kc << 7));
            u32x4 g0;
            g0[0] = 1u;                                   // count=1, user mode
            g0[1] = lds_base;                             // lds_addr
            g0[2] = (unsigned)(ga & 0xFFFFFFFFu);         // global_addr lo
            g0[3] = (unsigned)((ga >> 32) & 0x1FFFFFFu) | (2u << 30); // hi | type=2
            i32x8 g1;
            g1[0] = (1 << 16) | (1 << 20) | (5 << 22) | (3 << 25);
            //       data_size=2B  pad_en    pad_int=64dw  pad_amt=4dw
            g1[1] = (int)((512u & 0xFFFFu) << 16);        // tensor_dim0 = 512
            g1[2] = (int)(128u << 16);                    // tensor_dim1 = 128
            g1[3] = (int)(128u << 16);                    // tile_dim0   = 128
            g1[4] = 128;                                  // tile_dim1=128, tile_dim2=0
            g1[5] = 512;                                  // tensor_dim0_stride = 512
            g1[6] = 0;
            g1[7] = 0;
            i32x4 z4 = {};
#if __has_include(<hip/amd_detail/amd_gfx1250_TDM.h>)
            i32x8 z8 = {};
            __builtin_amdgcn_tensor_load_to_lds(g0, g1, z4, z4, z8, 0);
#else
            __builtin_amdgcn_tensor_load_to_lds(g0, g1, z4, z4, 0);
#endif
#if __has_builtin(__builtin_amdgcn_s_wait_tensorcnt)
            __builtin_amdgcn_s_wait_tensorcnt(0);
#else
            asm volatile("s_wait_tensorcnt 0x0" ::: "memory");
#endif
        }
#else
        // fallback: manual staging of the raw WattT chunk
        for (int i = tid; i < HATT * 128; i += 512) {
            int h = i >> 7, cc = i & 127;
            WqT[h * 136 + cc] = WattT[h * CD + (kc << 7) + cc];
        }
#endif
        __syncthreads();
#pragma unroll
        for (int ks = 0; ks < 4; ++ks) {
            const int k0 = (kc << 7) + (ks << 5);
            // A fragment = f16 feature row * gamma (packed f16 multiplies)
            const _Float16* pa = fTb + (m0 + l16) * CD + k0 + hi * 8;
            const _Float16* pg = gammaH + k0 + hi * 8;
            v8h alo = *(const v8h*)pa;
            v8h ahi = *(const v8h*)(pa + 16);
            v8h glo = *(const v8h*)pg;
            v8h ghi = *(const v8h*)(pg + 16);
            v16h af, gf;
#pragma unroll
            for (int j = 0; j < 8; ++j) {
                af[j] = alo[j];  af[8 + j] = ahi[j];
                gf[j] = glo[j];  gf[8 + j] = ghi[j];
            }
            af = af * gf;
#pragma unroll
            for (int nt = 0; nt < 8; ++nt) {
                const _Float16* pb = &WqT[(nt * 16 + l16) * 136 + (ks << 5) + hi * 8];
                v8h blo = *(const v8h*)pb;
                v8h bhi = *(const v8h*)(pb + 16);
                v16h bf;
#pragma unroll
                for (int j = 0; j < 8; ++j) { bf[j] = blo[j]; bf[8 + j] = bhi[j]; }
                acc[nt] = __builtin_amdgcn_wmma_f32_16x16x32_f16(
                    false, af, false, bf, (short)0, acc[nt], false, false);
            }
        }
    }

    // attention epilogue: gelu(hidden + cst) . W_att_f -> sigmoid
    float cv[8], wv[8];
#pragma unroll
    for (int nt = 0; nt < 8; ++nt) {
        int h = nt * 16 + l16;
        cv[nt] = cst[bq * HATT + h];
        wv[nt] = wfLds[h];
    }
    const float baf = b_att_f[0];
#pragma unroll
    for (int r = 0; r < 8; ++r) {
        float p = 0.f;
#pragma unroll
        for (int nt = 0; nt < 8; ++nt) {
            float v = acc[nt][r] + cv[nt];
            p += gelu_exact(v) * wv[nt];
        }
        p += __shfl_xor(p, 1);
        p += __shfl_xor(p, 2);
        p += __shfl_xor(p, 4);
        p += __shfl_xor(p, 8);
        if (l16 == 0) {
            int m = m0 + r + hi * 8;
            float a = 1.f / (1.f + expf(-(p + baf)));
            attnLds[m] = a;
            atomicAdd(&Ssum, a);
        }
    }
    __syncthreads();

    // pooling: one channel per thread; feature is (B, C, HW) so m is contiguous
    {
        const int c = tid;
        const float4* fp = (const float4*)(feature + (long)b * CD * HWN + (long)c * HWN);
        float s = 0.f;
        for (int m4 = 0; m4 < HWN / 4; ++m4) {
            float4 fv = fp[m4];
            s += attnLds[4 * m4 + 0] * fv.x + attnLds[4 * m4 + 1] * fv.y
               + attnLds[4 * m4 + 2] * fv.z + attnLds[4 * m4 + 3] * fv.w;
        }
        float S   = Ssum;
        float inv = 1.f / (S + 1e-8f);
        pooled[bq * CD + c] = gammaLds[c] * s * inv + betaLds[c] * S * inv;
    }
}

// ---------------- final tiny head: logits = H1 @ W_mlp2 + b2 ----------------
__global__ __launch_bounds__(64) void head2_kernel(
    const float* __restrict__ H1, const float* __restrict__ W2,
    const float* __restrict__ b2, float* __restrict__ out)
{
    __shared__ float red[64][NCLS];
    const int row = blockIdx.x, t = threadIdx.x;
    float acc[NCLS];
#pragma unroll
    for (int o = 0; o < NCLS; ++o) acc[o] = 0.f;
#pragma unroll
    for (int kk = 0; kk < 8; ++kk) {
        int k = t * 8 + kk;
        float h = H1[row * HMLP + k];
        const float* w = W2 + k * NCLS;
#pragma unroll
        for (int o = 0; o < NCLS; ++o) acc[o] += h * w[o];
    }
#pragma unroll
    for (int o = 0; o < NCLS; ++o) red[t][o] = acc[o];
    __syncthreads();
    if (t < NCLS) {
        float s = 0.f;
        for (int i = 0; i < 64; ++i) s += red[i][t];
        out[row * NCLS + t] = s + b2[t];
    }
}

// ---------------- launch ----------------
extern "C" void kernel_launch(void* const* d_in, const int* in_sizes, int n_in,
                              void* d_out, int out_size, void* d_ws, size_t ws_size,
                              hipStream_t stream) {
    const float* feature     = (const float*)d_in[0];
    const int*   label_ids   = (const int*)  d_in[1];
    const float* query_table = (const float*)d_in[2];
    const float* W_film      = (const float*)d_in[3];
    const float* b_film      = (const float*)d_in[4];
    const float* W_att_h     = (const float*)d_in[5];
    const float* b_att_h     = (const float*)d_in[6];
    const float* W_att_f     = (const float*)d_in[7];
    const float* b_att_f     = (const float*)d_in[8];
    const float* W_mlp1      = (const float*)d_in[9];
    const float* b_mlp1      = (const float*)d_in[10];
    const float* W_mlp2      = (const float*)d_in[11];
    const float* b_mlp2      = (const float*)d_in[12];
    float* out = (float*)d_out;

    char* ws = (char*)d_ws;
    _Float16* fT     = (_Float16*)(ws + 0);          // 2,097,152 B
    _Float16* WattT  = (_Float16*)(ws + 2097152);    //   131,072 B
    _Float16* W1T    = (_Float16*)(ws + 2228224);    //   524,288 B
    _Float16* WfilmT = (_Float16*)(ws + 2752512);    // 1,048,576 B
    float*    gb     = (float*)   (ws + 3801088);    // 2,097,152 B
    float*    cst    = (float*)   (ws + 5898240);    //   262,144 B
    float*    pooled = (float*)   (ws + 6160384);    // 1,048,576 B
    float*    H1     = (float*)   (ws + 7208960);    // 1,048,576 B

    // prep: transposed f16 copies
    tcvt_kernel<<<1024, 256, 0, stream>>>(feature, fT,     BB, CD, HWN);
    tcvt_kernel<<<256,  256, 0, stream>>>(W_att_h, WattT,  1,  CD, HATT);
    tcvt_kernel<<<512,  256, 0, stream>>>(W_mlp1,  W1T,    1,  CD, HMLP);
    tcvt_kernel<<<1024, 256, 0, stream>>>(W_film,  WfilmT, 1,  EE, 2 * CD);

    // gb = query_table[label_ids] @ W_film + b_film   (512 x 1024, K=512)
    gemm_f32a_f16b<<<256, 256, 0, stream>>>(query_table, EE, label_ids,
                                            WfilmT, EE, b_film, gb, 1024, 64, 0);
    // cst = beta @ W_att_h + b_att_h                  (512 x 128, K=512)
    gemm_f32a_f16b<<<32, 256, 0, stream>>>(gb + 512, 1024, nullptr,
                                           WattT, CD, b_att_h, cst, HATT, 8, 0);
    // FiLM attention + pooling (the 17 GFLOP part)
    attn_pool_kernel<<<BQ, 512, 0, stream>>>(fT, WattT, feature, gb, cst,
                                             W_att_f, b_att_f, pooled);
    // H1 = gelu(pooled @ W_mlp1 + b_mlp1)             (512 x 512, K=512)
    gemm_f32a_f16b<<<128, 256, 0, stream>>>(pooled, CD, nullptr,
                                            W1T, CD, b_mlp1, H1, HMLP, 32, 1);
    // logits = H1 @ W_mlp2 + b_mlp2
    head2_kernel<<<BQ, 64, 0, stream>>>(H1, W_mlp2, b_mlp2, out);
}
